// TrajectoryDecoder_15118284882789
// MI455X (gfx1250) — matrix operand order repaired and checked by reading, not compiled, not run on hardware
//
#include <hip/hip_runtime.h>

typedef __attribute__((ext_vector_type(16))) _Float16 v16h;
typedef __attribute__((ext_vector_type(8)))  _Float16 v8h;
typedef __attribute__((ext_vector_type(8)))  float    v8f;
typedef __attribute__((ext_vector_type(4)))  float    v4f;

#define CTX_D   256
#define H_D     128
#define ENC_D   128
#define T_STEPS 128
#define GATES   512      // 4*H
#define GSTR    516      // padded row stride (floats) to dodge LDS bank conflicts
#define BATCH   8192
#define ROWS    16       // batch rows per block
#define NTHREADS 128     // 4 waves
#define PPN     (ROWS * 8 * 2)

__device__ __forceinline__ float fast_tanh(float x) {
#if __has_builtin(__builtin_amdgcn_tanhf)
    return __builtin_amdgcn_tanhf(x);          // v_tanh_f32 (gfx1250 TRANS unit)
#else
    const float e = __expf(2.0f * x);
    return 1.0f - 2.0f / (e + 1.0f);
#endif
}
__device__ __forceinline__ float sigm(float x) { return 0.5f + 0.5f * fast_tanh(0.5f * x); }

__global__ __launch_bounds__(NTHREADS, 1)
void lstm_traj_decoder(const float* __restrict__ ctx,
                       const float* __restrict__ enc,
                       const float* __restrict__ ball,
                       const float* __restrict__ Wh,  const float* __restrict__ bh,
                       const float* __restrict__ Wc,  const float* __restrict__ bc,
                       const float* __restrict__ Wih, const float* __restrict__ Whh,
                       const float* __restrict__ bih, const float* __restrict__ bhh,
                       const float* __restrict__ Wo,  const float* __restrict__ bo,
                       float* __restrict__ out)
{
    // gbuf: holds gate_static_eff during preamble, then per-step gates (padded rows)
    __shared__ float gbuf[ROWS * GSTR];                 // ~33 KB
    __shared__ v16h  afragv[4 * 32];                    // h as f16 A-fragments, 4 KB
    __shared__ float ppart[2 * PPN];                    // double-buffered pred partials
    __shared__ float prevb[ROWS * 2];                   // pred(h0) for step-0 correction

    v8h* afrag8 = (v8h*)afragv;

    const int t    = threadIdx.x;
    const int lane = t & 31;
    const int wave = t >> 5;          // 0..3 -> gate group i/f/g/o
    const int n    = lane & 15;       // N column within 16-wide tile
    const int half = lane >> 4;       // C/D row-half selector
    const int m    = t & 15;          // elementwise: batch row in block
    const int seg  = t >> 4;          // elementwise: hidden segment (0..7)
    const int u0   = seg * 16;
    const int ktE  = u0 >> 5;         // A-fragment K-tile this thread writes
    const int jsel = seg & 1;         // jstart/8 for the two 16B h-runs
    const int row0 = blockIdx.x * ROWS;

    const float bo0 = bo[0], bo1 = bo[1];

    // W_prev = W_ih[:, :2] for this lane's 8 gate columns (IN_DIM = 132)
    float wp0[8], wp1[8];
    #pragma unroll
    for (int jt = 0; jt < 8; ++jt) {
        const int j = (wave * 8 + jt) * 16 + n;
        wp0[jt] = Wih[j * 132 + 0];
        wp1[jt] = Wih[j * 132 + 1];
    }

    // ---- W_eff = W_hh + W_prev @ Wo  -> resident WMMA B-fragments (f16) ------
    // B 32x16 layout: lane holds column N = lane&15, elems j -> K = 8*(lane>>4) + j + (j&8).
    v16h wb[8][4];
    #pragma unroll
    for (int jt = 0; jt < 8; ++jt) {
        const int jcol = (wave * 8 + jt) * 16 + n;
        #pragma unroll
        for (int kt = 0; kt < 4; ++kt) {
            const int k0 = kt * 32 + half * 8;
            const float* src = Whh + jcol * H_D + k0;
            const float* wo0p = Wo + k0;
            const float* wo1p = Wo + H_D + k0;
            v16h b;
            #pragma unroll
            for (int j = 0; j < 16; ++j)
                b[j] = (_Float16)(src[j + (j & 8)] + wp0[jt] * wo0p[j + (j & 8)] + wp1[jt] * wo1p[j + (j & 8)]);
            wb[jt][kt] = b;
        }
    }

    // Wo slices for the 16 hidden units this thread owns elementwise
    float wo0[16], wo1[16];
    #pragma unroll
    for (int j = 0; j < 16; ++j) { wo0[j] = Wo[u0 + j]; wo1[j] = Wo[H_D + u0 + j]; }

    // ---------------- preamble: h0, c0 and pred(h0) partials ------------------
    const float* cxr = ctx + (size_t)(row0 + m) * CTX_D;
    float cst[16];                                   // c state lives in registers
    {
        _Float16 hb[16];
        float pi0 = 0.0f, pi1 = 0.0f;
        #pragma unroll 1
        for (int j = 0; j < 16; ++j) {
            const int u = u0 + j;
            const float* whr = Wh + u * CTX_D;
            const float* wcr = Wc + u * CTX_D;
            float ah = bh[u], ac = bc[u];
            for (int k = 0; k < CTX_D; k += 4) {
                v4f x  = *(const v4f*)(cxr + k);
                v4f w1 = *(const v4f*)(whr + k);
                v4f w2 = *(const v4f*)(wcr + k);
                ah += x[0]*w1[0] + x[1]*w1[1] + x[2]*w1[2] + x[3]*w1[3];
                ac += x[0]*w2[0] + x[1]*w2[1] + x[2]*w2[2] + x[3]*w2[3];
            }
            cst[j] = ac;
            hb[j]  = (_Float16)ah;
            pi0 += ah * wo0[j];
            pi1 += ah * wo1[j];
        }
        v8h lo, hi;
        #pragma unroll
        for (int i = 0; i < 8; ++i) { lo[i] = hb[i]; hi[i] = hb[8 + i]; }
        afrag8[(ktE * 32 + m) * 2 + jsel]      = lo;
        afrag8[(ktE * 32 + 16 + m) * 2 + jsel] = hi;
        ppart[(m * 8 + seg) * 2 + 0] = pi0;
        ppart[(m * 8 + seg) * 2 + 1] = pi1;
    }

    {   // gate_static_eff = static@W_stat.T + b_ih + b_hh + W_prev@bo
        const float bx = ball[(row0 + m) * 2 + 0];
        const float by = ball[(row0 + m) * 2 + 1];
        const float* er = enc + (size_t)(row0 + m) * ENC_D;
        #pragma unroll 1
        for (int jj = 0; jj < 64; ++jj) {
            const int j = seg * 64 + jj;
            const float* wr = Wih + j * 132;
            float a = bih[j] + bhh[j] + wr[0] * bo0 + wr[1] * bo1
                      + bx * wr[2] + by * wr[3];
            for (int k = 0; k < ENC_D; k += 4) {
                v4f x = *(const v4f*)(er + k);
                v4f w = *(const v4f*)(wr + 4 + k);
                a += x[0]*w[0] + x[1]*w[1] + x[2]*w[2] + x[3]*w[3];
            }
            gbuf[m * GSTR + j] = a;
        }
    }
    __syncthreads();

    if (t < ROWS) {     // pred(h0) -> prevb (step-0 correction input)
        float p0 = bo0, p1 = bo1;
        #pragma unroll
        for (int s = 0; s < 8; ++s) {
            p0 += ppart[(t * 8 + s) * 2 + 0];
            p1 += ppart[(t * 8 + s) * 2 + 1];
        }
        prevb[t * 2 + 0] = p0;
        prevb[t * 2 + 1] = p1;
    }

    // gate_static_eff tiles -> resident C-seed VGPRs (c[r] = [r+8*half][jt*16+n])
    v8f gsv[8];
    #pragma unroll
    for (int jt = 0; jt < 8; ++jt) {
        #pragma unroll
        for (int r = 0; r < 8; ++r)
            gsv[jt][r] = gbuf[(r + 8 * half) * GSTR + (wave * 8 + jt) * 16 + n];
    }
    __syncthreads();   // gbuf now the per-step gates buffer

    // ---- phases ----
    auto tile_phase = [&](bool first) {
        v16h av[4];
        #pragma unroll
        for (int kt = 0; kt < 4; ++kt) av[kt] = afragv[kt * 32 + lane];
        float p0c[8], p1c[8];
        if (first) {
            #pragma unroll
            for (int r = 0; r < 8; ++r) {
                p0c[r] = prevb[(r + 8 * half) * 2 + 0];
                p1c[r] = prevb[(r + 8 * half) * 2 + 1];
            }
        }
        #pragma unroll
        for (int jt = 0; jt < 8; ++jt) {
            v8f acc = gsv[jt];
            if (first) {   // undo the folded prev-term at step 0 (prev0 == 0)
                #pragma unroll
                for (int r = 0; r < 8; ++r)
                    acc[r] -= p0c[r] * wp0[jt] + p1c[r] * wp1[jt];
            }
            #pragma unroll
            for (int kt = 0; kt < 4; ++kt)
                acc = __builtin_amdgcn_wmma_f32_16x16x32_f16(
                        false, av[kt], false, wb[jt][kt], (short)0, acc, false, false);
            #pragma unroll
            for (int r = 0; r < 8; ++r)
                gbuf[(r + 8 * half) * GSTR + (wave * 8 + jt) * 16 + n] = acc[r];
        }
    };

    auto cell_phase = [&](int step) {
        // deferred pred reduction for the previous step (off the critical path)
        if (step > 0 && t < ROWS) {
            const float* pp = ppart + ((step - 1) & 1) * PPN;
            float p0 = bo0, p1 = bo1;
            #pragma unroll
            for (int s = 0; s < 8; ++s) {
                p0 += pp[(t * 8 + s) * 2 + 0];
                p1 += pp[(t * 8 + s) * 2 + 1];
            }
            float* op = out + ((size_t)(row0 + t) * T_STEPS + (step - 1)) * 2;
            op[0] = p0;
            op[1] = p1;
        }
        // LSTM cell on (m, u0..u0+15); c in registers
        const float* gm = gbuf + m * GSTR;
        _Float16 hb[16];
        float pw0 = 0.0f, pw1 = 0.0f;
        #pragma unroll
        for (int q = 0; q < 4; ++q) {
            v4f gi = *(const v4f*)(gm + u0 + 4 * q);
            v4f gf = *(const v4f*)(gm + 128 + u0 + 4 * q);
            v4f gg = *(const v4f*)(gm + 256 + u0 + 4 * q);
            v4f go = *(const v4f*)(gm + 384 + u0 + 4 * q);
            #pragma unroll
            for (int e = 0; e < 4; ++e) {
                const int jj = 4 * q + e;
                const float cc = sigm(gf[e]) * cst[jj] + sigm(gi[e]) * fast_tanh(gg[e]);
                cst[jj] = cc;
                const float h = sigm(go[e]) * fast_tanh(cc);
                hb[jj] = (_Float16)h;
                pw0 += h * wo0[jj];
                pw1 += h * wo1[jj];
            }
        }
        v8h lo, hi;
        #pragma unroll
        for (int i = 0; i < 8; ++i) { lo[i] = hb[i]; hi[i] = hb[8 + i]; }
        afrag8[(ktE * 32 + m) * 2 + jsel]      = lo;
        afrag8[(ktE * 32 + 16 + m) * 2 + jsel] = hi;
        float* pp = ppart + (step & 1) * PPN;
        pp[(m * 8 + seg) * 2 + 0] = pw0;
        pp[(m * 8 + seg) * 2 + 1] = pw1;
    };

    // ------------------------------ recurrence --------------------------------
    tile_phase(true);
    __syncthreads();
    #pragma unroll 1
    for (int step = 0; step < T_STEPS - 1; ++step) {
        cell_phase(step);
        __syncthreads();
        tile_phase(false);
        __syncthreads();
    }
    cell_phase(T_STEPS - 1);
    __syncthreads();

    if (t < ROWS) {   // final pred
        const float* pp = ppart + ((T_STEPS - 1) & 1) * PPN;
        float p0 = bo0, p1 = bo1;
        #pragma unroll
        for (int s = 0; s < 8; ++s) {
            p0 += pp[(t * 8 + s) * 2 + 0];
            p1 += pp[(t * 8 + s) * 2 + 1];
        }
        float* op = out + ((size_t)(row0 + t) * T_STEPS + (T_STEPS - 1)) * 2;
        op[0] = p0;
        op[1] = p1;
    }
}

extern "C" void kernel_launch(void* const* d_in, const int* in_sizes, int n_in,
                              void* d_out, int out_size, void* d_ws, size_t ws_size,
                              hipStream_t stream) {
    (void)in_sizes; (void)n_in; (void)d_ws; (void)ws_size; (void)out_size;
    const float* ctx  = (const float*)d_in[0];
    const float* enc  = (const float*)d_in[1];
    const float* ball = (const float*)d_in[2];
    // d_in[3] = max_frames (int scalar) == 128, baked into T_STEPS
    const float* Wh  = (const float*)d_in[4];
    const float* bh  = (const float*)d_in[5];
    const float* Wc  = (const float*)d_in[6];
    const float* bc  = (const float*)d_in[7];
    const float* Wih = (const float*)d_in[8];
    const float* Whh = (const float*)d_in[9];
    const float* bih = (const float*)d_in[10];
    const float* bhh = (const float*)d_in[11];
    const float* Wo  = (const float*)d_in[12];
    const float* bo  = (const float*)d_in[13];
    float* out = (float*)d_out;

    dim3 grid(BATCH / ROWS);
    dim3 block(NTHREADS);
    lstm_traj_decoder<<<grid, block, 0, stream>>>(ctx, enc, ball, Wh, bh, Wc, bc,
                                                  Wih, Whh, bih, bhh, Wo, bo, out);
}
